// FastSelfAttention_81544249082585
// MI455X (gfx1250) — matrix operand order repaired and checked by reading, not compiled, not run on hardware
//
#include <hip/hip_runtime.h>
#include <hip/hip_bf16.h>

#define B_ 8
#define L_ 2048
#define D_ 1024
#define H_ 16
#define K_ 31
#define DC_ 64
#define M_ (B_ * L_)
#define NCHUNK 16
#define CLEN (L_ / NCHUNK) /* 128 */

typedef __attribute__((ext_vector_type(16))) __bf16 v16bf;
typedef __attribute__((ext_vector_type(8)))  float  v8f;

union FragB { v16bf v; uint4 q[2]; };

__device__ __forceinline__ unsigned short f2bf(float f) {
  unsigned int u = __float_as_uint(f);
  u += 0x7fffu + ((u >> 16) & 1u);
  return (unsigned short)(u >> 16);
}

// ---------------- K0: W_fc fp32 -> bf16 ----------------
__global__ __launch_bounds__(256) void k0_cvt_w(const float* __restrict__ W,
                                                unsigned short* __restrict__ Wb) {
  int idx = blockIdx.x * 256 + threadIdx.x;          // 0 .. 262143
  float4 v = reinterpret_cast<const float4*>(W)[idx];
  int o = idx * 4;
  Wb[o + 0] = f2bf(v.x); Wb[o + 1] = f2bf(v.y);
  Wb[o + 2] = f2bf(v.z); Wb[o + 3] = f2bf(v.w);
}

// ---------------- K1a: per-chunk sums ----------------
__global__ __launch_bounds__(256) void k1a_chunksum(const float* __restrict__ x,
                                                    float* __restrict__ chunkS) {
  int g = blockIdx.x * 256 + threadIdx.x;            // B*NCHUNK*D threads
  int d = g & (D_ - 1);
  int bc = g >> 10;
  int c = bc & (NCHUNK - 1);
  int b = bc >> 4;
  const float* xp = x + ((size_t)(b * L_ + c * CLEN) * D_ + d);
  float s = 0.f;
  for (int i = 0; i < CLEN; ++i) s += xp[(size_t)i * D_];
  chunkS[g] = s;
}

// ---------------- K1b: scan within chunk -> prefix mean ----------------
__global__ __launch_bounds__(256) void k1b_scan(const float* __restrict__ x,
                                                const float* __restrict__ chunkS,
                                                float* __restrict__ avg) {
  int g = blockIdx.x * 256 + threadIdx.x;
  int d = g & (D_ - 1);
  int bc = g >> 10;
  int c = bc & (NCHUNK - 1);
  int b = bc >> 4;
  float acc = 0.f;
  for (int cc = 0; cc < c; ++cc) acc += chunkS[((b * NCHUNK + cc) << 10) + d];
  size_t base = (size_t)(b * L_ + c * CLEN) * D_ + d;
  const float* xp = x + base;
  float* ap = avg + base;
  for (int i = 0; i < CLEN; ++i) {
    acc += xp[(size_t)i * D_];
    ap[(size_t)i * D_] = acc / (float)(c * CLEN + i + 1);
  }
}

// ---------------- K2: projection + gate + softmax(+mask after) ----------------
__global__ __launch_bounds__(256) void k2_weights(const float* __restrict__ avg,
                                                  const float* __restrict__ Wpe,
                                                  const float* __restrict__ bpe,
                                                  const float* __restrict__ stt,
                                                  float* __restrict__ wbuf) {
  __shared__ float Ws[DC_ * 2 * K_];   // 64 x 62
  __shared__ float bs[2 * K_];
  __shared__ float ss[H_ * K_];
  int tid = threadIdx.x;
  for (int i = tid; i < DC_ * 2 * K_; i += 256) Ws[i] = Wpe[i];
  if (tid < 2 * K_) bs[tid] = bpe[tid];
  for (int i = tid; i < H_ * K_; i += 256) ss[i] = stt[i];
  __syncthreads();

  int p = blockIdx.x * 256 + tid;          // position index over B*H*L
  int l = p & (L_ - 1);
  int bh = p >> 11;
  int h = bh & (H_ - 1);
  int b = bh >> 4;

  const float* cip = avg + ((size_t)(b * L_ + l) * D_ + h * DC_);
  float ci[DC_];
  #pragma unroll
  for (int q = 0; q < DC_ / 4; ++q)
    reinterpret_cast<float4*>(ci)[q] = reinterpret_cast<const float4*>(cip)[q];

  float en[K_];
  float mx = -1e30f;
  #pragma unroll 1
  for (int j = 0; j < K_; ++j) {
    float dsum = bs[j], gsum = bs[j + K_];
    #pragma unroll
    for (int k = 0; k < DC_; ++k) {
      float c = ci[k];
      dsum += c * Ws[k * (2 * K_) + j];
      gsum += c * Ws[k * (2 * K_) + j + K_];
    }
    float sg = 1.0f / (1.0f + __expf(-gsum));
    float e = ss[h * K_ + j] + sg * dsum;
    en[j] = e;
    mx = fmaxf(mx, e);
  }
  float s = 0.f;
  #pragma unroll
  for (int j = 0; j < K_; ++j) { en[j] = __expf(en[j] - mx); s += en[j]; }
  float inv = 1.0f / s;
  float* wp = wbuf + (size_t)p * 32;
  #pragma unroll
  for (int j = 0; j < K_; ++j) wp[j] = (l + j >= K_ - 1) ? en[j] * inv : 0.f;
  wp[31] = 0.f;
}

// ---------------- K3: causal windowed conv -> bf16 ----------------
__global__ __launch_bounds__(256) void k3_conv(const float* __restrict__ avg,
                                               const float* __restrict__ wbuf,
                                               unsigned short* __restrict__ conv) {
  __shared__ float ciS[(32 + K_ - 1) * DC_];   // 62 x 64
  __shared__ float wS[32 * 32];
  int tid = threadIdx.x;
  int bh = blockIdx.x >> 6;
  int lt = blockIdx.x & 63;
  int l0 = lt * 32;
  int h = bh & (H_ - 1);
  int b = bh >> 4;

  // stage avg window rows [l0-30 .. l0+31]
  for (int i = tid; i < (32 + K_ - 1) * DC_; i += 256) {
    int rr = i >> 6;
    int d = i & 63;
    int t = l0 - (K_ - 1) + rr;
    ciS[i] = (t >= 0) ? avg[(size_t)(b * L_ + t) * D_ + h * DC_ + d] : 0.f;
  }
  // stage weights
  for (int i = tid; i < 32 * 32; i += 256) {
    int li = i >> 5;
    int k = i & 31;
    wS[i] = wbuf[(size_t)(bh * L_ + l0 + li) * 32 + k];
  }
  __syncthreads();

  int d = tid & 63;
  #pragma unroll
  for (int i = 0; i < 8; ++i) {
    int li = (tid >> 6) * 8 + i;
    float a = 0.f;
    #pragma unroll
    for (int k = 0; k < K_; ++k) a += wS[li * 32 + k] * ciS[(li + k) * DC_ + d];
    conv[(size_t)(b * L_ + l0 + li) * D_ + h * DC_ + d] = f2bf(a);
  }
}

// ---------------- K4: WMMA GEMM + residual + LayerNorm (fused) ----------------
__global__ __launch_bounds__(256) void k4_gemm_ln(const unsigned short* __restrict__ Abf,
                                                  const unsigned short* __restrict__ Wbf,
                                                  const float* __restrict__ xres,
                                                  const float* __restrict__ gamma,
                                                  const float* __restrict__ beta,
                                                  float* __restrict__ out) {
  __shared__ float wsum[8][16];
  __shared__ float wsq[8][16];
  __shared__ float muS[16];
  __shared__ float rsS[16];

  const int tid = threadIdx.x;
  const int lane = tid & 31;
  const int wave = tid >> 5;
  const int laneN = lane & 15;
  const int m0 = blockIdx.x * 16;
  const int rowAdd = (lane & 16) ? 8 : 0;
  const int kbaseA = (lane & 16) ? 8 : 0;    // A: lanes 16-31 hold K {8..15,24..31}
  const int kbaseB = (lane & 16) ? 8 : 0;    // B: lanes 16-31 hold K {8..15,24..31}

  const unsigned short* arow = Abf + (size_t)(m0 + laneN) * D_;

  v8f acc[8];
  const v8f zero = {0.f, 0.f, 0.f, 0.f, 0.f, 0.f, 0.f, 0.f};
  #pragma unroll
  for (int t = 0; t < 8; ++t) acc[t] = zero;

  for (int kt = 0; kt < D_; kt += 32) {
    FragB a;
    const uint4* ap = reinterpret_cast<const uint4*>(arow + kt + kbaseA);
    a.q[0] = ap[0];      // K = kt+kbaseA .. +7
    a.q[1] = ap[2];      // K = kt+kbaseA+16 .. +23
    #pragma unroll
    for (int t = 0; t < 8; ++t) {
      int n = wave * 128 + t * 16 + laneN;
      const uint4* bp = reinterpret_cast<const uint4*>(Wbf + (size_t)n * D_ + kt + kbaseB);
      FragB bf;
      bf.q[0] = bp[0];   // K = kt+kbaseB .. +7
      bf.q[1] = bp[2];   // K = kt+kbaseB+16 .. +23
      acc[t] = __builtin_amdgcn_wmma_f32_16x16x32_bf16(
          false, a.v, false, bf.v, (short)0, acc[t], false, false);
    }
  }

  // residual add
  #pragma unroll
  for (int t = 0; t < 8; ++t) {
    int n = wave * 128 + t * 16 + laneN;
    #pragma unroll
    for (int r = 0; r < 8; ++r)
      acc[t][r] += xres[(size_t)(m0 + rowAdd + r) * D_ + n];
  }

  // per-lane per-row partial sums over this lane's 1 column-per-tile x 8 tiles
  float ps[8], pq[8];
  #pragma unroll
  for (int r = 0; r < 8; ++r) { ps[r] = 0.f; pq[r] = 0.f; }
  #pragma unroll
  for (int t = 0; t < 8; ++t) {
    #pragma unroll
    for (int r = 0; r < 8; ++r) { float v = acc[t][r]; ps[r] += v; pq[r] += v * v; }
  }
  // reduce across the 16 lanes sharing each row half
  #pragma unroll
  for (int off = 1; off < 16; off <<= 1) {
    #pragma unroll
    for (int r = 0; r < 8; ++r) {
      ps[r] += __shfl_xor(ps[r], off, 32);
      pq[r] += __shfl_xor(pq[r], off, 32);
    }
  }
  if (laneN == 0) {   // lanes 0 and 16
    #pragma unroll
    for (int r = 0; r < 8; ++r) {
      wsum[wave][rowAdd + r] = ps[r];
      wsq[wave][rowAdd + r] = pq[r];
    }
  }
  __syncthreads();
  if (tid < 16) {
    float ss = 0.f, qq = 0.f;
    #pragma unroll
    for (int wv = 0; wv < 8; ++wv) { ss += wsum[wv][tid]; qq += wsq[wv][tid]; }
    float mu = ss * (1.0f / (float)D_);
    float var = fmaxf(qq * (1.0f / (float)D_) - mu * mu, 0.f);
    muS[tid] = mu;
    rsS[tid] = rsqrtf(var + 1e-6f);
  }
  __syncthreads();

  #pragma unroll
  for (int t = 0; t < 8; ++t) {
    int n = wave * 128 + t * 16 + laneN;
    float gm = gamma[n], bt = beta[n];
    #pragma unroll
    for (int r = 0; r < 8; ++r) {
      int row = rowAdd + r;
      float v = (acc[t][r] - muS[row]) * rsS[row] * gm + bt;
      out[(size_t)(m0 + row) * D_ + n] = v;
    }
  }
}

// ---------------- launch ----------------
extern "C" void kernel_launch(void* const* d_in, const int* in_sizes, int n_in,
                              void* d_out, int out_size, void* d_ws, size_t ws_size,
                              hipStream_t stream) {
  const float* x     = (const float*)d_in[0];
  const float* W_pe  = (const float*)d_in[1];
  const float* b_pe  = (const float*)d_in[2];
  const float* stt   = (const float*)d_in[3];
  const float* W_fc  = (const float*)d_in[4];
  const float* gamma = (const float*)d_in[5];
  const float* beta  = (const float*)d_in[6];
  float* out = (float*)d_out;

  char* ws = (char*)d_ws;
  const size_t OFF_AVG   = 0;                               // B*L*D f32   = 64 MB
  const size_t OFF_CHUNK = OFF_AVG + (size_t)M_ * D_ * 4;   // B*16*D f32  = 512 KB
  const size_t OFF_W     = OFF_CHUNK + (size_t)B_ * NCHUNK * D_ * 4;  // B*H*L*32 f32 = 32 MB
  const size_t OFF_CONV  = OFF_W + (size_t)B_ * H_ * L_ * 32 * 4;     // B*L*D bf16 = 32 MB
  const size_t OFF_WBF   = OFF_CONV + (size_t)M_ * D_ * 2;            // D*D bf16 = 2 MB

  float* avg            = (float*)(ws + OFF_AVG);
  float* chunkS         = (float*)(ws + OFF_CHUNK);
  float* wbuf           = (float*)(ws + OFF_W);
  unsigned short* conv  = (unsigned short*)(ws + OFF_CONV);
  unsigned short* Wbf   = (unsigned short*)(ws + OFF_WBF);

  k0_cvt_w<<<(D_ * D_) / (256 * 4), 256, 0, stream>>>(W_fc, Wbf);
  k1a_chunksum<<<(B_ * NCHUNK * D_) / 256, 256, 0, stream>>>(x, chunkS);
  k1b_scan<<<(B_ * NCHUNK * D_) / 256, 256, 0, stream>>>(x, chunkS, avg);
  k2_weights<<<(B_ * H_ * L_) / 256, 256, 0, stream>>>(avg, W_pe, b_pe, stt, wbuf);
  k3_conv<<<B_ * H_ * (L_ / 32), 256, 0, stream>>>(avg, wbuf, conv);
  k4_gemm_ln<<<M_ / 16, 256, 0, stream>>>(conv, Wbf, x, gamma, beta, out);
}
